// LanguageModel_14156212207627
// MI455X (gfx1250) — matrix operand order repaired and checked by reading, not compiled, not run on hardware
//
#include <hip/hip_runtime.h>
#include <hip/hip_bf16.h>
#include <math.h>

// ---------------------------------------------------------------------------
// LanguageModel forward for MI455X (gfx1250, wave32, WMMA).
//   tokens[16,4096] -> embedding-bag(<=4 subtok) -> window-sum(8) ->
//   relu(x@W1+b1) -> relu(@W2+b2) -> @Wout+bout -> log_softmax
// Heavy GEMMs run on v_wmma_f32_16x16x32_bf16 with f32 accumulation.
// Wave tile 32Mx64N (8 accumulators) -> 0.75 KB fragment traffic per WMMA.
// ---------------------------------------------------------------------------

typedef __bf16 bf16;
typedef __bf16  v16bf __attribute__((ext_vector_type(16)));
typedef float   v8f   __attribute__((ext_vector_type(8)));

union Frag16 {            // 16 bf16 values = 8 VGPRs, loaded as 2x b128
    v16bf v;
    uint4 q[2];
};

#define VOCAB     32000
#define EMBED_DIM 128
#define HIDDEN    512
#define MAX_SUB   4
#define CTX       8
#define ROWS      8192            // B*S/CTX = 16*4096/8
#define NTOK      65536           // B*S

// ---------------------------------------------------------------------------
// 1) f32 [K,N] -> bf16 [N,K] tiled transpose+convert (coalesced both sides).
// ---------------------------------------------------------------------------
__global__ __launch_bounds__(256)
void transpose_cvt_kernel(const float* __restrict__ in, bf16* __restrict__ out,
                          int K, int N)
{
    __shared__ float tile[32][33];
    const int n0 = blockIdx.x * 32;
    const int k0 = blockIdx.y * 32;
    const int tx = threadIdx.x;           // 0..31
    for (int j = threadIdx.y; j < 32; j += 8)
        tile[j][tx] = in[(size_t)(k0 + j) * N + (n0 + tx)];
    __syncthreads();
    for (int j = threadIdx.y; j < 32; j += 8)
        out[(size_t)(n0 + j) * K + (k0 + tx)] = (bf16)tile[tx][j];
}

// ---------------------------------------------------------------------------
// 2) Fused ragged embedding-bag + context-window(8) sum -> win bf16 [8192,128]
//    One block per output row, one thread per embedding dim.
//    Token/subtoken indices are block-uniform -> compiler emits scalar loads.
// ---------------------------------------------------------------------------
__global__ __launch_bounds__(128)
void embed_window_kernel(const int* __restrict__ tokens,
                         const int* __restrict__ subtok_table,
                         const float* __restrict__ subtok_mask,
                         const float* __restrict__ embed_table,
                         bf16* __restrict__ win)
{
    const int row = blockIdx.x;          // 0..8191
    const int d   = threadIdx.x;         // 0..127
    const int t0  = row * CTX;
    float acc = 0.0f;
#pragma unroll
    for (int j = 0; j < CTX; ++j) {
        const int tok = tokens[t0 + j];
#pragma unroll
        for (int m = 0; m < MAX_SUB; ++m) {
            const int   sid = subtok_table[tok * MAX_SUB + m];
            const float msk = subtok_mask[tok * MAX_SUB + m];
            acc += embed_table[(size_t)sid * EMBED_DIM + d] * msk;
        }
    }
    win[(size_t)row * EMBED_DIM + d] = (bf16)acc;
}

// ---------------------------------------------------------------------------
// 3) WMMA GEMM:  C[M,N] = act(A[M,K] @ Bt[N,K]^T + bias)
//    A row-major bf16, B pre-transposed row-major bf16 (so both fragment
//    loads are two contiguous b128 per lane, matching the 16-bit A 16x32
//    and B 32x16 VGPR layouts: lanes 0-15 / 16-31 split the K halves).
//    Block: 256 thr = 8 waves arranged 2(M) x 4(N); block tile 64M x 256N;
//    each wave computes a 32x64 tile = 8 accumulators of v8f.
//    Per K-step: 12 b128 loads for 8 WMMAs (0.75 KB fragment traffic/WMMA).
//    MODE 0: ReLU, bf16 output.  MODE 1: f32 output (logits).
// ---------------------------------------------------------------------------
template <int MODE>
__global__ __launch_bounds__(256)
void gemm_wmma_kernel(const bf16* __restrict__ A,
                      const bf16* __restrict__ Bt,
                      const float* __restrict__ bias,
                      void* __restrict__ Cout,
                      int M, int N, int K)
{
    const int tid  = threadIdx.x;
    const int w    = tid >> 5;
    const int lane = tid & 31;
    const int wm   = w & 1;              // 0..1  (M)
    const int wn   = w >> 1;             // 0..3  (N)
    const int mBase = blockIdx.y * 64  + wm * 32;
    const int nBase = blockIdx.x * 256 + wn * 64;
    const int lm = lane & 15;            // row/col within 16
    const int lh = lane >> 4;            // K-half select

    v8f acc[2][4] = {};

    const bf16* aRow = A  + (size_t)(mBase + lm) * K;
    const bf16* bRow = Bt + (size_t)(nBase + lm) * K;

    for (int k = 0; k < K; k += 32) {
        Frag16 a[2], b[4];
#pragma unroll
        for (int mi = 0; mi < 2; ++mi) {
            const bf16* p = aRow + (size_t)mi * 16 * K + k;
            // A 16x32 bf16: lanes 0-15 hold K {0..7,16..23}, lanes 16-31 {8..15,24..31}
            a[mi].q[0] = *reinterpret_cast<const uint4*>(p + lh * 8);
            a[mi].q[1] = *reinterpret_cast<const uint4*>(p + 16 + lh * 8);
        }
#pragma unroll
        for (int ni = 0; ni < 4; ++ni) {
            const bf16* p = bRow + (size_t)ni * 16 * K + k;
            // B 32x16 bf16: lanes 0-15 hold K {0..7,16..23}, lanes 16-31 {8..15,24..31}
            b[ni].q[0] = *reinterpret_cast<const uint4*>(p + lh * 8);
            b[ni].q[1] = *reinterpret_cast<const uint4*>(p + 16 + lh * 8);
        }
        if (k + 32 < K) {                 // hint next K-slice into cache
            __builtin_prefetch(aRow + k + 32, 0, 3);
            __builtin_prefetch(bRow + k + 32, 0, 3);
        }
#pragma unroll
        for (int mi = 0; mi < 2; ++mi)
#pragma unroll
            for (int ni = 0; ni < 4; ++ni)
                acc[mi][ni] = __builtin_amdgcn_wmma_f32_16x16x32_bf16(
                    false, a[mi].v, false, b[ni].v,
                    (short)0, acc[mi][ni], false, false);
    }

    // Epilogue. C/D layout: lane 0-15 -> N=lane, VGPR r -> M=r (+8 for lanes 16-31).
    const int colL = nBase + lm;
    const int rowL = mBase + lh * 8;
#pragma unroll
    for (int mi = 0; mi < 2; ++mi) {
#pragma unroll
        for (int ni = 0; ni < 4; ++ni) {
            const int col = colL + ni * 16;
            const float bv = bias[col];
#pragma unroll
            for (int r = 0; r < 8; ++r) {
                const int row = rowL + mi * 16 + r;
                float v = acc[mi][ni][r] + bv;
                if (MODE == 0) {
                    v = fmaxf(v, 0.0f);
                    ((bf16*)Cout)[(size_t)row * N + col] = (bf16)v;
                } else {
                    ((float*)Cout)[(size_t)row * N + col] = v;
                }
            }
        }
    }
}

// ---------------------------------------------------------------------------
// 4) Per-row online logsumexp over the 32000-wide logits (single read pass).
// ---------------------------------------------------------------------------
__global__ __launch_bounds__(256)
void row_lse_kernel(const float* __restrict__ logits,
                    float* __restrict__ lse, int N)
{
    __shared__ float sm[256];
    __shared__ float ss[256];
    const int row = blockIdx.x;
    const int tid = threadIdx.x;
    const float* p = logits + (size_t)row * N;

    float m = -INFINITY, s = 0.0f;
    for (int j = tid; j < N; j += 256) {
        const float x = p[j];
        if (x > m) { s *= __expf(m - x); m = x; }
        s += __expf(x - m);
    }
    sm[tid] = m; ss[tid] = s;
    __syncthreads();
    for (int off = 128; off > 0; off >>= 1) {
        if (tid < off) {
            const float m1 = sm[tid], s1 = ss[tid];
            const float m2 = sm[tid + off], s2 = ss[tid + off];
            const float mm = fmaxf(m1, m2);
            ss[tid] = s1 * __expf(m1 - mm) + s2 * __expf(m2 - mm);
            sm[tid] = mm;
        }
        __syncthreads();
    }
    if (tid == 0) lse[row] = sm[0] + logf(ss[0]);
}

// ---------------------------------------------------------------------------
// 5) out = logit - lse[row]
// ---------------------------------------------------------------------------
__global__ __launch_bounds__(256)
void sub_lse_kernel(float* __restrict__ out, const float* __restrict__ lse, int N)
{
    const int row = blockIdx.y;
    const int col = blockIdx.x * 256 + threadIdx.x;
    out[(size_t)row * N + col] -= lse[row];
}

// ---------------------------------------------------------------------------
extern "C" void kernel_launch(void* const* d_in, const int* in_sizes, int n_in,
                              void* d_out, int out_size, void* d_ws, size_t ws_size,
                              hipStream_t stream)
{
    (void)in_sizes; (void)n_in; (void)out_size; (void)ws_size;

    const int*   tokens       = (const int*)  d_in[0];
    const int*   subtok_table = (const int*)  d_in[1];
    const float* subtok_mask  = (const float*)d_in[2];
    const float* embed_table  = (const float*)d_in[3];
    const float* W1   = (const float*)d_in[4];
    const float* b1   = (const float*)d_in[5];
    const float* W2   = (const float*)d_in[6];
    const float* b2   = (const float*)d_in[7];
    const float* Wout = (const float*)d_in[8];
    const float* bout = (const float*)d_in[9];
    float* out = (float*)d_out;

    // Workspace carve-out (256B aligned slabs), ~52 MB total.
    char*  ws  = (char*)d_ws;
    size_t off = 0;
    auto carve = [&](size_t bytes) -> void* {
        void* p = ws + off;
        off += (bytes + 255) & ~(size_t)255;
        return p;
    };
    bf16*  win   = (bf16*) carve((size_t)ROWS * EMBED_DIM * sizeof(bf16));
    bf16*  W1t   = (bf16*) carve((size_t)HIDDEN * EMBED_DIM * sizeof(bf16));
    bf16*  W2t   = (bf16*) carve((size_t)HIDDEN * HIDDEN * sizeof(bf16));
    bf16*  Woutt = (bf16*) carve((size_t)VOCAB * HIDDEN * sizeof(bf16));
    bf16*  h1    = (bf16*) carve((size_t)ROWS * HIDDEN * sizeof(bf16));
    bf16*  h2    = (bf16*) carve((size_t)ROWS * HIDDEN * sizeof(bf16));
    float* lse   = (float*)carve((size_t)ROWS * sizeof(float));

    // Weight convert+transpose: f32 [K,N] -> bf16 [N,K]
    transpose_cvt_kernel<<<dim3(HIDDEN / 32, EMBED_DIM / 32), dim3(32, 8), 0, stream>>>(
        W1, W1t, EMBED_DIM, HIDDEN);
    transpose_cvt_kernel<<<dim3(HIDDEN / 32, HIDDEN / 32), dim3(32, 8), 0, stream>>>(
        W2, W2t, HIDDEN, HIDDEN);
    transpose_cvt_kernel<<<dim3(VOCAB / 32, HIDDEN / 32), dim3(32, 8), 0, stream>>>(
        Wout, Woutt, HIDDEN, VOCAB);

    // Embedding-bag + window sum -> win bf16 [8192,128]
    embed_window_kernel<<<ROWS, EMBED_DIM, 0, stream>>>(
        tokens, subtok_table, subtok_mask, embed_table, win);

    // h1 = relu(win @ W1 + b1)                 [8192,512]
    gemm_wmma_kernel<0><<<dim3(HIDDEN / 256, ROWS / 64), 256, 0, stream>>>(
        win, W1t, b1, h1, ROWS, HIDDEN, EMBED_DIM);
    // h2 = relu(h1 @ W2 + b2)                  [8192,512]
    gemm_wmma_kernel<0><<<dim3(HIDDEN / 256, ROWS / 64), 256, 0, stream>>>(
        h1, W2t, b2, h2, ROWS, HIDDEN, HIDDEN);
    // logits = h2 @ Wout + bout -> d_out       [8192,32000]
    gemm_wmma_kernel<1><<<dim3(VOCAB / 256, ROWS / 64), 256, 0, stream>>>(
        h2, Woutt, bout, out, ROWS, VOCAB, HIDDEN);

    // log_softmax: per-row logsumexp then subtract.
    row_lse_kernel<<<ROWS, 256, 0, stream>>>(out, lse, VOCAB);
    sub_lse_kernel<<<dim3(VOCAB / 256, ROWS), 256, 0, stream>>>(out, lse, VOCAB);
}
